// SubsequencAware_LSTM_11879879544319
// MI455X (gfx1250) — matrix operand order repaired and checked by reading, not compiled, not run on hardware
//
#include <hip/hip_runtime.h>
#include <hip/hip_bf16.h>
#include <cstdint>

// ---------------- types & problem constants ----------------
typedef __bf16 bf16;
typedef __attribute__((ext_vector_type(16))) __bf16 v16bf;
typedef __attribute__((ext_vector_type(8)))  float  v8f;

#define B_SZ 256
#define T_SZ 512
#define D_SZ 128
#define H_SZ 512
#define NR   2049          // 4H+1 meaningful output columns
#define NP   2112          // padded N = 33 * 64
#define K1   1152          // 512 + 128 + 512
#define K2   1024          // 512 + 512

union Frag32 { uint4 u[2]; v16bf v; };

__device__ __forceinline__ float sigm(float x) { return 1.0f / (1.0f + __expf(-x)); }

// ---------------- weight packing (f32 -> bf16, padded rows zeroed) ----------------
__global__ void pack_w1(const float* __restrict__ U11, const float* __restrict__ W01,
                        const float* __restrict__ U21, bf16* __restrict__ Wp) {
    int idx = blockIdx.x * blockDim.x + threadIdx.x;   // NP * K1 threads
    int n = idx / K1, k = idx % K1;
    float v = 0.0f;
    if (n < NR) {
        if (k < 512)       v = U11[n * 512 + k];
        else if (k < 640)  v = W01[n * 128 + (k - 512)];
        else               v = U21[n * 512 + (k - 640)];
    }
    Wp[idx] = (bf16)v;
}

__global__ void pack_w2(const float* __restrict__ U11, const float* __restrict__ W01,
                        bf16* __restrict__ Wp) {
    int idx = blockIdx.x * blockDim.x + threadIdx.x;   // NP * K2 threads
    int n = idx / K2, k = idx % K2;
    float v = 0.0f;
    if (n < NR) {
        if (k < 512) v = U11[n * 512 + k];
        else         v = W01[n * 512 + (k - 512)];
    }
    Wp[idx] = (bf16)v;
}

// ---------------- per-step scaled-concat input builders ----------------
__global__ void prep1(const float* __restrict__ h1, const float* __restrict__ h2,
                      const float* __restrict__ xt,   // inputs + t*D ; row stride T*D
                      const float* __restrict__ z1o, bf16* __restrict__ S1) {
    int idx = blockIdx.x * blockDim.x + threadIdx.x;   // B * K1 threads
    int b = idx / K1, k = idx % K1;
    float z = z1o[b];
    float v;
    if (k < 512)       v = (1.0f - z) * h1[b * H_SZ + k];
    else if (k < 640)  v = xt[b * (T_SZ * D_SZ) + (k - 512)];
    else               v = z * h2[b * H_SZ + (k - 640)];
    S1[idx] = (bf16)v;
}

__global__ void prep2(const float* __restrict__ h1, const float* __restrict__ h2,
                      const float* __restrict__ z2o, const float* __restrict__ z1n,
                      bf16* __restrict__ S2) {
    int idx = blockIdx.x * blockDim.x + threadIdx.x;   // B * K2 threads
    int b = idx / K2, k = idx % K2;
    float v;
    if (k < 512) v = (1.0f - z2o[b]) * h2[b * H_SZ + k];
    else         v = z1n[b] * h1[b * H_SZ + (k - 512)];
    S2[idx] = (bf16)v;
}

// ---------------- WMMA GEMM: (256 x K) * (NP x K)^T -> (256 x NP) f32 ----------------
// grid = 66 blocks x 256 threads -> 528 waves = 16 m-tiles x 33 n-strips (16x64 each)
__global__ __launch_bounds__(256) void gemm_bf16(const bf16* __restrict__ Sm,
                                                 const bf16* __restrict__ Wp,
                                                 float* __restrict__ out, int K) {
    const int lane = threadIdx.x & 31;
    const int wave = threadIdx.x >> 5;
    const int gw   = blockIdx.x * 8 + wave;       // 0..527
    const int m0   = (gw / 33) * 16;
    const int n0   = (gw % 33) * 64;

    v8f acc0 = {}, acc1 = {}, acc2 = {}, acc3 = {};

    // A fragment addressing (16x32 bf16): lane l%16 = row, lane half selects K {0..15}/{16..31}
    const bf16* aPtr = Sm + (m0 + (lane & 15)) * K + ((lane >> 4) << 4);
    // B fragment addressing (32x16 bf16): lane l%16 = col (weight row), lane half selects K {0..15}/{16..31}
    const int bOff = ((lane >> 4) << 4);
    const bf16* b0 = Wp + (n0 +  0 + (lane & 15)) * K + bOff;
    const bf16* b1 = Wp + (n0 + 16 + (lane & 15)) * K + bOff;
    const bf16* b2 = Wp + (n0 + 32 + (lane & 15)) * K + bOff;
    const bf16* b3 = Wp + (n0 + 48 + (lane & 15)) * K + bOff;

    for (int k = 0; k < K; k += 32) {
        __builtin_prefetch(aPtr + k + 128, 0, 1);
        __builtin_prefetch(b0 + k + 128, 0, 1);

        Frag32 a;
        a.u[0] = *(const uint4*)(aPtr + k);        // K chunk +0..7  (this half)
        a.u[1] = *(const uint4*)(aPtr + k + 8);    // K chunk +8..15 (this half)

        Frag32 fb0, fb1, fb2, fb3;
        fb0.u[0] = *(const uint4*)(b0 + k); fb0.u[1] = *(const uint4*)(b0 + k + 8);
        fb1.u[0] = *(const uint4*)(b1 + k); fb1.u[1] = *(const uint4*)(b1 + k + 8);
        fb2.u[0] = *(const uint4*)(b2 + k); fb2.u[1] = *(const uint4*)(b2 + k + 8);
        fb3.u[0] = *(const uint4*)(b3 + k); fb3.u[1] = *(const uint4*)(b3 + k + 8);

        acc0 = __builtin_amdgcn_wmma_f32_16x16x32_bf16(false, a.v, false, fb0.v, (short)0, acc0, false, false);
        acc1 = __builtin_amdgcn_wmma_f32_16x16x32_bf16(false, a.v, false, fb1.v, (short)0, acc1, false, false);
        acc2 = __builtin_amdgcn_wmma_f32_16x16x32_bf16(false, a.v, false, fb2.v, (short)0, acc2, false, false);
        acc3 = __builtin_amdgcn_wmma_f32_16x16x32_bf16(false, a.v, false, fb3.v, (short)0, acc3, false, false);
    }

    // C/D layout: lanes 0-15: N=lane, M = m0 + r; lanes 16-31: N=lane-16, M = m0 + 8 + r
    const int rBase = m0 + ((lane >> 4) << 3);
    const int cCol  = lane & 15;
#pragma unroll
    for (int r = 0; r < 8; ++r) {
        float* row = out + (rBase + r) * NP + n0 + cCol;
        row[0]  = acc0[r];
        row[16] = acc1[r];
        row[32] = acc2[r];
        row[48] = acc3[r];
    }
}

// ---------------- LSTM cell elementwise update ----------------
__global__ void cell_update(const float* __restrict__ s, const float* __restrict__ bias,
                            float* __restrict__ c, float* __restrict__ h,
                            const float* __restrict__ z_old, const float* __restrict__ zb_arr,
                            float* __restrict__ z_new,
                            float* __restrict__ y, int yStride) {
    int idx = blockIdx.x * blockDim.x + threadIdx.x;   // B * H threads
    int b = idx >> 9;          // / 512
    int j = idx & (H_SZ - 1);

    const float* srow = s + b * NP;
    float fg = sigm(srow[j]            + bias[j]);
    float ig = sigm(srow[H_SZ + j]     + bias[H_SZ + j]);
    float og = sigm(srow[2 * H_SZ + j] + bias[2 * H_SZ + j]);
    float gg = tanhf(srow[3 * H_SZ + j] + bias[3 * H_SZ + j]);

    float zh = (srow[4 * H_SZ] + bias[4 * H_SZ] + 1.0f) * 0.5f;   // hard_sigm, a=1
    zh = fminf(fmaxf(zh, 0.0f), 1.0f);

    float z  = z_old[b];
    float zb = zb_arr ? zb_arr[b] : 1.0f;

    float igg = ig * gg;
    float co  = c[idx];
    float cn  = z * igg + (1.0f - z) * (1.0f - zb) * co
                        + (1.0f - z) * zb * (fg * co + igg);
    float tc  = tanhf(cn);
    float ho  = h[idx];
    float ot  = og * tc;
    float hn  = z * ot + (1.0f - z) * (1.0f - zb) * ho + (1.0f - z) * zb * ot;

    c[idx] = cn;
    h[idx] = hn;
    if (y) y[b * yStride + j] = hn * zh;   // output uses the NEW z of this step
    if (j == 0) z_new[b] = zh;
}

// ---------------- host orchestration ----------------
extern "C" void kernel_launch(void* const* d_in, const int* in_sizes, int n_in,
                              void* d_out, int out_size, void* d_ws, size_t ws_size,
                              hipStream_t stream) {
    (void)in_sizes; (void)n_in; (void)out_size; (void)ws_size;

    const float* inputs = (const float*)d_in[0];  // (B,T,D)
    const float* U11_1  = (const float*)d_in[1];  // (2049,512)
    const float* U21_1  = (const float*)d_in[2];  // (2049,512)
    const float* W01_1  = (const float*)d_in[3];  // (2049,128)
    const float* b1     = (const float*)d_in[4];  // (2049)
    const float* U11_2  = (const float*)d_in[5];  // (2049,512)
    const float* W01_2  = (const float*)d_in[6];  // (2049,512)
    const float* b2     = (const float*)d_in[7];  // (2049)
    float* out = (float*)d_out;                   // (B,T,H2)

    // workspace carve-out (all offsets are multiples of 256B)
    char* ws = (char*)d_ws;
    size_t off = 0;
    auto take = [&](size_t bytes) { char* p = ws + off; off += (bytes + 255) & ~size_t(255); return p; };
    bf16*  Wp1 = (bf16*)take((size_t)NP * K1 * 2);
    bf16*  Wp2 = (bf16*)take((size_t)NP * K2 * 2);
    bf16*  S1  = (bf16*)take((size_t)B_SZ * K1 * 2);
    bf16*  S2  = (bf16*)take((size_t)B_SZ * K2 * 2);
    float* sb1 = (float*)take((size_t)B_SZ * NP * 4);
    float* sb2 = (float*)take((size_t)B_SZ * NP * 4);
    float* states = (float*)take((size_t)4 * B_SZ * H_SZ * 4 + 4 * B_SZ * 4);
    float* h1 = states;
    float* c1 = h1 + B_SZ * H_SZ;
    float* h2 = c1 + B_SZ * H_SZ;
    float* c2 = h2 + B_SZ * H_SZ;
    float* z1p[2] = { c2 + 4LL * B_SZ * H_SZ - 3LL * B_SZ * H_SZ + 3LL * B_SZ * H_SZ, nullptr };
    // (compute z pointers plainly)
    float* zbase = c2 + B_SZ * H_SZ;
    z1p[0] = zbase;            z1p[1] = zbase + B_SZ;
    float* z2p[2] = { zbase + 2 * B_SZ, zbase + 3 * B_SZ };

    // zero all recurrent state (h, c, z ping-pong buffers)
    hipMemsetAsync(states, 0, (size_t)4 * B_SZ * H_SZ * 4 + 4 * B_SZ * 4, stream);

    // pack weights to bf16 (once per launch; deterministic)
    pack_w1<<<(NP * K1) / 256, 256, 0, stream>>>(U11_1, W01_1, U21_1, Wp1);
    pack_w2<<<(NP * K2) / 256, 256, 0, stream>>>(U11_2, W01_2, Wp2);

    for (int t = 0; t < T_SZ; ++t) {
        float* z1_old = z1p[t & 1];
        float* z1_new = z1p[(t + 1) & 1];
        float* z2_old = z2p[t & 1];
        float* z2_new = z2p[(t + 1) & 1];

        // ---- layer 1 ----
        prep1<<<(B_SZ * K1) / 256, 256, 0, stream>>>(h1, h2, inputs + (size_t)t * D_SZ, z1_old, S1);
        gemm_bf16<<<66, 256, 0, stream>>>(S1, Wp1, sb1, K1);
        cell_update<<<(B_SZ * H_SZ) / 256, 256, 0, stream>>>(
            sb1, b1, c1, h1, z1_old, /*zb=*/nullptr, z1_new, /*y=*/nullptr, 0);

        // ---- layer 2 ----
        prep2<<<(B_SZ * K2) / 256, 256, 0, stream>>>(h1, h2, z2_old, z1_new, S2);
        gemm_bf16<<<66, 256, 0, stream>>>(S2, Wp2, sb2, K2);
        cell_update<<<(B_SZ * H_SZ) / 256, 256, 0, stream>>>(
            sb2, b2, c2, h2, z2_old, /*zb=*/z1_new, z2_new,
            /*y=*/out + (size_t)t * H_SZ, /*yStride=*/T_SZ * H_SZ);
    }
}
